// MHLAttention_1443109011474
// MI455X (gfx1250) — matrix operand order repaired and checked by reading, not compiled, not run on hardware
//
#include <hip/hip_runtime.h>

typedef _Float16 half_t;
typedef __attribute__((ext_vector_type(16))) _Float16 v16h;
typedef __attribute__((ext_vector_type(8)))  _Float16 v8h;
typedef __attribute__((ext_vector_type(8)))  float    v8f;

#define WMMA_F32_F16(a, b, c) \
  __builtin_amdgcn_wmma_f32_16x16x32_f16(false, (a), false, (b), (short)0, (c), false, false)

__device__ __forceinline__ v16h pack16(v8h lo, v8h hi) {
  v16h r;
#pragma unroll
  for (int i = 0; i < 8; ++i) { r[i] = lo[i]; r[i + 8] = hi[i]; }
  return r;
}

// LDS byte address from a generic (__shared__) pointer: the flat address of an
// LDS location is {SHARED_BASE, offset[31:0]}, so truncation yields the LDS
// offset used by async-to-LDS VDST operands.
__device__ __forceinline__ uint32_t lds_addr_of(const void* p) {
  return (uint32_t)(uintptr_t)p;
}

// Async global->LDS 16B copy (gfx1250 GLOBAL_LOAD_ASYNC_TO_LDS_B128, ASYNCcnt).
__device__ __forceinline__ void async_copy16(uint32_t lds_byte_addr, const void* gaddr) {
  asm volatile("global_load_async_to_lds_b128 %0, %1, off"
               :: "v"(lds_byte_addr), "v"(gaddr)
               : "memory");
}
__device__ __forceinline__ void wait_async0() {
  asm volatile("s_wait_asynccnt 0x0" ::: "memory");
}

// ---------------------------------------------------------------------------
// fp32 -> fp16 elementwise convert
// ---------------------------------------------------------------------------
__global__ void f32_to_f16(const float* __restrict__ in, half_t* __restrict__ out, int n) {
  int i = blockIdx.x * blockDim.x + threadIdx.x;
  int stride = gridDim.x * blockDim.x;
  for (; i < n; i += stride) out[i] = (half_t)in[i];
}

// ---------------------------------------------------------------------------
// Generic WMMA GEMM: C[MxN] = A[MxK] * B[KxN], A/B f16 row-major, C = CT
// Block = 128 threads (4 waves), tile 64x64, k-step 32.
// ---------------------------------------------------------------------------
template <typename CT>
__global__ __launch_bounds__(128) void gemm_wmma(const half_t* __restrict__ A,
                                                 const half_t* __restrict__ B,
                                                 CT* __restrict__ C,
                                                 int M, int N, int K) {
  __shared__ alignas(16) half_t As[64][32];
  __shared__ alignas(16) half_t Bs[64][32];  // [n][k]

  const int tid  = threadIdx.x;
  const int wave = tid >> 5;
  const int lane = tid & 31;
  const int ln   = lane & 15;
  const int lg   = lane >> 4;
  const int m0   = blockIdx.y * 64;
  const int n0   = blockIdx.x * 64;

  v8f acc[4] = {};

  for (int k0 = 0; k0 < K; k0 += 32) {
    __syncthreads();
    // A tile: 64 rows x 64 bytes = 256 x 16B chunks
#pragma unroll
    for (int c = tid; c < 256; c += 128) {
      int row = c >> 2, cp = c & 3;
      *(float4*)(&As[row][cp * 8]) =
          *(const float4*)(A + (size_t)(m0 + row) * K + k0 + cp * 8);
    }
    // B tile (transposed into LDS): 32 k-rows x 64 cols as 1024 u32 pairs
    for (int i = tid; i < 1024; i += 128) {
      int k = i >> 5, n = (i & 31) << 1;
      union { uint32_t u; half_t x[2]; } cv;
      cv.u = *(const uint32_t*)(B + (size_t)(k0 + k) * N + n0 + n);
      Bs[n][k]     = cv.x[0];
      Bs[n + 1][k] = cv.x[1];
    }
    // Prefetch next k-tile while this one is consumed (global_prefetch_b8).
    if (k0 + 32 < K) {
      __builtin_prefetch(A + (size_t)(m0 + (tid >> 1)) * K + (k0 + 32) + (tid & 1) * 16, 0, 0);
      __builtin_prefetch(B + (size_t)(k0 + 32 + (tid >> 2)) * N + n0 + (tid & 3) * 16, 0, 0);
    }
    __syncthreads();

    const int arow = wave * 16 + ln;
    v16h af = pack16(*(const v8h*)&As[arow][lg * 8],
                     *(const v8h*)&As[arow][lg * 8 + 16]);
#pragma unroll
    for (int ns = 0; ns < 4; ++ns) {
      const half_t* bp = &Bs[ns * 16 + ln][lg * 8];
      v16h bf = pack16(*(const v8h*)bp, *(const v8h*)(bp + 16));
      acc[ns] = WMMA_F32_F16(af, bf, acc[ns]);
    }
  }

  const int rowb = m0 + wave * 16 + lg * 8;
#pragma unroll
  for (int ns = 0; ns < 4; ++ns) {
    const int col = n0 + ns * 16 + ln;
#pragma unroll
    for (int r = 0; r < 8; ++r) {
      C[(size_t)(rowb + r) * N + col] = (CT)acc[ns][r];
    }
  }
}

// ---------------------------------------------------------------------------
// RMSNorm over rows of 192 fp32, output fp16. One row per 64-thread block.
// ---------------------------------------------------------------------------
__global__ __launch_bounds__(64) void rmsnorm_f16(const float* __restrict__ H,
                                                  const float* __restrict__ w,
                                                  half_t* __restrict__ out) {
  const int row = blockIdx.x;
  const int tid = threadIdx.x;
  float v[3];
  float s = 0.f;
#pragma unroll
  for (int i = 0; i < 3; ++i) {
    v[i] = H[(size_t)row * 192 + tid + i * 64];
    s += v[i] * v[i];
  }
#pragma unroll
  for (int off = 16; off >= 1; off >>= 1) s += __shfl_xor(s, off, 32);
  __shared__ float red[2];
  if ((tid & 31) == 0) red[tid >> 5] = s;
  __syncthreads();
  const float scale = rsqrtf((red[0] + red[1]) * (1.0f / 192.0f) + 1e-6f);
#pragma unroll
  for (int i = 0; i < 3; ++i) {
    const int c = tid + i * 64;
    out[(size_t)row * 192 + c] = (half_t)(v[i] * scale * w[c]);
  }
}

// ---------------------------------------------------------------------------
// RoPE + repack qkv (B,S,9216 head-minor d*16+h) -> q,k,v each (B*H, S, 192)
// ---------------------------------------------------------------------------
__global__ __launch_bounds__(192) void rope_pack(const half_t* __restrict__ qkv,
                                                 half_t* __restrict__ q,
                                                 half_t* __restrict__ k,
                                                 half_t* __restrict__ v) {
  const int S  = 2048;
  const int bs = blockIdx.x;       // b*S + s
  const int b  = bs >> 11;
  const int s  = bs & 2047;
  const int d  = threadIdx.x;
  const bool rot = (d >= 128) && (d < 160);
  float c = 1.f, sn = 0.f, sgn = 0.f;
  if (rot) {
    const int p = (d - 128) >> 1;
    const float inv = __expf(-(float)p * 0.5756462732485114f);  // ln(1e4)/16
    const float fr = (float)s * inv;
    c = __cosf(fr);
    sn = __sinf(fr);
    sgn = (d & 1) ? 1.f : -1.f;
  }
  const size_t base = (size_t)bs * 9216;
#pragma unroll
  for (int h = 0; h < 16; ++h) {
    const size_t orow = ((size_t)(b * 16 + h) * S + s) * 192 + d;
    float xq = (float)qkv[base + d * 16 + h];
    float xk = (float)qkv[base + 3072 + d * 16 + h];
    float xv = (float)qkv[base + 6144 + d * 16 + h];
    if (rot) {
      const float pq = (float)qkv[base + (d ^ 1) * 16 + h];
      const float pk = (float)qkv[base + 3072 + (d ^ 1) * 16 + h];
      xq = xq * c + sgn * pq * sn;
      xk = xk * c + sgn * pk * sn;
    }
    q[orow] = (half_t)xq;
    k[orow] = (half_t)xk;
    v[orow] = (half_t)xv;
  }
}

// ---------------------------------------------------------------------------
// Flash attention per (b,h): causal, online softmax in fp32.
// Grid (B*H, S/64), block 128 (4 waves). Wave = 16 query rows.
// K tile async-copied to LDS once per block (GLOBAL_LOAD_ASYNC_TO_LDS_B128);
// V staged transposed in LDS; Q fragments resident in VGPRs.
// ---------------------------------------------------------------------------
__global__ __launch_bounds__(128) void mhla_attn(const half_t* __restrict__ Q,
                                                 const half_t* __restrict__ K,
                                                 const half_t* __restrict__ V,
                                                 half_t* __restrict__ O) {
  const int S = 2048, DH = 192;
  const int bh = blockIdx.x;
  const int b = bh >> 4, h = bh & 15;
  const int q0 = blockIdx.y * 64;
  const int tid  = threadIdx.x;
  const int wave = tid >> 5;
  const int lane = tid & 31;
  const int ln = lane & 15, lg = lane >> 4;

  const half_t* Qb = Q + (size_t)bh * S * DH;
  const half_t* Kb = K + (size_t)bh * S * DH;
  const half_t* Vb = V + (size_t)bh * S * DH;

  __shared__ alignas(16) half_t Ks[64 * 192];    // K tile (linear copy)  24KB
  __shared__ alignas(16) half_t Vs[192][64];     // V^T tile [d][key]     24KB
  __shared__ alignas(16) half_t Ps[4][16][64];   // per-wave P tiles       8KB

  // Q A-fragments for this wave's 16 rows, all 6 k-steps (d = 0..191)
  v16h qf[6];
  {
    const half_t* qp = Qb + (size_t)(q0 + wave * 16 + ln) * DH;
#pragma unroll
    for (int ks = 0; ks < 6; ++ks) {
      const int kb = ks * 32 + lg * 8;
      qf[ks] = pack16(*(const v8h*)(qp + kb), *(const v8h*)(qp + kb + 16));
    }
  }

  v8f o[12] = {};
  float mrow[8], lrow[8];
#pragma unroll
  for (int r = 0; r < 8; ++r) { mrow[r] = -3.0e38f; lrow[r] = 0.f; }
  const float scale = 0.07216878364870322f;  // 1/sqrt(192)

  const uint32_t ks_base = lds_addr_of(&Ks[0]);

  for (int j0 = 0; j0 <= q0; j0 += 64) {
    __syncthreads();  // all waves done with previous Ks/Vs

    // Async copy K tile (contiguous 24576B) global -> LDS.
    {
      const char* ksrc = (const char*)(Kb + (size_t)j0 * DH);
#pragma unroll
      for (int i = 0; i < 12; ++i) {
        const uint32_t off = (uint32_t)(tid + i * 128) * 16u;
        async_copy16(ks_base + off, ksrc + off);
      }
    }
    // Stage V tile transposed: Vs[d][j] = V[j0+j][d]
    for (int i = tid; i < 64 * 96; i += 128) {
      const int jr = i / 96;
      const int dp = (i - jr * 96) * 2;
      union { uint32_t u; half_t x[2]; } cv;
      cv.u = *(const uint32_t*)(Vb + (size_t)(j0 + jr) * DH + dp);
      Vs[dp][jr]     = cv.x[0];
      Vs[dp + 1][jr] = cv.x[1];
    }
    wait_async0();
    __syncthreads();  // Ks written, Vs stores visible

    // Prefetch next tiles into cache while computing this one.
    if (j0 + 64 <= q0) {
      __builtin_prefetch((const char*)(Kb + (size_t)(j0 + 64) * DH) + tid * 192, 0, 0);
      __builtin_prefetch((const char*)(Vb + (size_t)(j0 + 64) * DH) + tid * 192, 0, 0);
    }

    // S = Q K^T  (B fragments from LDS: Ks[key][d], d contiguous)
    v8f sacc[4] = {};
#pragma unroll
    for (int ks = 0; ks < 6; ++ks) {
      const int kb = ks * 32 + lg * 8;
#pragma unroll
      for (int ns = 0; ns < 4; ++ns) {
        const half_t* kp = &Ks[(size_t)(ns * 16 + ln) * 192 + kb];
        v16h bf = pack16(*(const v8h*)kp, *(const v8h*)(kp + 16));
        sacc[ns] = WMMA_F32_F16(qf[ks], bf, sacc[ns]);
      }
    }

    // Online softmax (fp32). Row for VGPR r = q0 + wave*16 + lg*8 + r.
    float corr[8];
#pragma unroll
    for (int r = 0; r < 8; ++r) {
      const int grow = q0 + wave * 16 + lg * 8 + r;
      float mx = -3.0e38f;
#pragma unroll
      for (int ns = 0; ns < 4; ++ns) {
        float val = sacc[ns][r] * scale;
        if (j0 + ns * 16 + ln > grow) val = -1.0e9f;  // causal mask
        sacc[ns][r] = val;
        mx = fmaxf(mx, val);
      }
#pragma unroll
      for (int msk = 1; msk < 16; msk <<= 1) mx = fmaxf(mx, __shfl_xor(mx, msk, 16));
      const float mnew = fmaxf(mrow[r], mx);
      corr[r] = __expf(mrow[r] - mnew);
      float psum = 0.f;
#pragma unroll
      for (int ns = 0; ns < 4; ++ns) {
        const float p = __expf(sacc[ns][r] - mnew);
        sacc[ns][r] = p;
        psum += p;
      }
#pragma unroll
      for (int msk = 1; msk < 16; msk <<= 1) psum += __shfl_xor(psum, msk, 16);
      lrow[r] = lrow[r] * corr[r] + psum;
      mrow[r] = mnew;
    }
#pragma unroll
    for (int ns2 = 0; ns2 < 12; ++ns2)
#pragma unroll
      for (int r = 0; r < 8; ++r) o[ns2][r] *= corr[r];

    // P (C layout) -> per-wave LDS, then reload as A fragments.
#pragma unroll
    for (int ns = 0; ns < 4; ++ns)
#pragma unroll
      for (int r = 0; r < 8; ++r)
        Ps[wave][lg * 8 + r][ns * 16 + ln] = (half_t)sacc[ns][r];

    // O += P @ V   (kdim = 64 keys -> 2 steps)
#pragma unroll
    for (int ks2 = 0; ks2 < 2; ++ks2) {
      const half_t* pp = &Ps[wave][ln][ks2 * 32 + lg * 8];
      v16h pf = pack16(*(const v8h*)pp, *(const v8h*)(pp + 16));
#pragma unroll
      for (int ns2 = 0; ns2 < 12; ++ns2) {
        const half_t* vp = &Vs[ns2 * 16 + ln][ks2 * 32 + lg * 8];
        v16h bf = pack16(*(const v8h*)vp, *(const v8h*)(vp + 16));
        o[ns2] = WMMA_F32_F16(pf, bf, o[ns2]);
      }
    }
  }

  // Epilogue: normalize and store to (b, s, h*192 + d) as fp16.
  const int rowb = q0 + wave * 16 + lg * 8;
#pragma unroll
  for (int r = 0; r < 8; ++r) {
    const float inv = 1.0f / lrow[r];
    const size_t obase = ((size_t)(b * S + rowb + r)) * 3072 + h * 192 + ln;
#pragma unroll
    for (int ns2 = 0; ns2 < 12; ++ns2)
      O[obase + ns2 * 16] = (half_t)(o[ns2][r] * inv);
  }
}

// ---------------------------------------------------------------------------
extern "C" void kernel_launch(void* const* d_in, const int* in_sizes, int n_in,
                              void* d_out, int out_size, void* d_ws, size_t ws_size,
                              hipStream_t stream) {
  (void)in_sizes; (void)n_in; (void)out_size; (void)ws_size;
  const float* x      = (const float*)d_in[0];
  // d_in[1] = mask (implicit causal; unused)
  const float* w_down = (const float*)d_in[2];
  const float* rms_w  = (const float*)d_in[3];
  const float* w_up   = (const float*)d_in[4];
  const float* w_o    = (const float*)d_in[5];
  float* out = (float*)d_out;

  char* ws = (char*)d_ws;
  size_t off = 0;
  auto alloc = [&](size_t bytes) -> char* {
    char* p = ws + off;
    off += (bytes + 255) & ~(size_t)255;
    return p;
  };
  half_t* x16   = (half_t*)alloc((size_t)4194304 * 2);   // (B,S,1024)
  half_t* wd16  = (half_t*)alloc((size_t)196608 * 2);    // (1024,192)
  half_t* wu16  = (half_t*)alloc((size_t)1769472 * 2);   // (192,9216)
  half_t* wo16  = (half_t*)alloc((size_t)3145728 * 2);   // (3072,1024)
  float*  h32   = (float*) alloc((size_t)786432 * 4);    // (4096,192)
  half_t* h16   = (half_t*)alloc((size_t)786432 * 2);
  half_t* qkv16 = (half_t*)alloc((size_t)37748736 * 2);  // (4096,9216)
  half_t* q16   = (half_t*)alloc((size_t)12582912 * 2);  // (B*H,S,192)
  half_t* k16   = (half_t*)alloc((size_t)12582912 * 2);
  half_t* v16   = (half_t*)alloc((size_t)12582912 * 2);
  half_t* ao16  = (half_t*)alloc((size_t)12582912 * 2);  // (4096,3072)

  f32_to_f16<<<2048, 256, 0, stream>>>(x, x16, 4194304);
  f32_to_f16<<<192, 256, 0, stream>>>(w_down, wd16, 196608);
  f32_to_f16<<<1024, 256, 0, stream>>>(w_up, wu16, 1769472);
  f32_to_f16<<<1536, 256, 0, stream>>>(w_o, wo16, 3145728);

  // h = x @ w_down : (4096,1024)x(1024,192)
  gemm_wmma<float><<<dim3(3, 64), 128, 0, stream>>>(x16, wd16, h32, 4096, 192, 1024);
  rmsnorm_f16<<<4096, 64, 0, stream>>>(h32, rms_w, h16);
  // qkv = h @ w_up : (4096,192)x(192,9216)
  gemm_wmma<half_t><<<dim3(144, 64), 128, 0, stream>>>(h16, wu16, qkv16, 4096, 9216, 192);
  rope_pack<<<4096, 192, 0, stream>>>(qkv16, q16, k16, v16);
  // attention: 32 (b,h) pairs x 32 query tiles
  mhla_attn<<<dim3(32, 32), 128, 0, stream>>>(q16, k16, v16, ao16);
  // out = attn_out @ w_o : (4096,3072)x(3072,1024) -> fp32 d_out
  gemm_wmma<float><<<dim3(16, 64), 128, 0, stream>>>(ao16, wo16, out, 4096, 1024, 3072);
}
